// Transformer_32083405701836
// MI455X (gfx1250) — matrix operand order repaired and checked by reading, not compiled, not run on hardware
//
#include <hip/hip_runtime.h>

// ---------------------------------------------------------------------------
// Transformer block for MI455X (gfx1250, wave32, WMMA + Tensor Data Mover).
// Mixed precision: bf16 operands, f32 accumulation via v_wmma_f32_16x16x32_bf16.
// Tile staging uses TDM (tensor_load_to_lds, TENSORcnt) with double-buffered
// LDS when the builtin is available; falls back to manual VGPR staging.
// ---------------------------------------------------------------------------

#define NHEADS 8
#define HEADSZ 64
#define DMODEL 512
#define SEQ    2048
#define BATCH  4
#define KSIZE  2048
#define MROWS  (BATCH * SEQ)   // 8192

typedef __attribute__((ext_vector_type(16))) __bf16 v16bf;
typedef __attribute__((ext_vector_type(8)))  float  v8f;
typedef unsigned short u16;

#if defined(__HIP_DEVICE_COMPILE__) && \
    __has_builtin(__builtin_amdgcn_tensor_load_to_lds) && \
    __has_builtin(__builtin_amdgcn_s_wait_tensorcnt)
#define HAVE_TDM 1
#else
#define HAVE_TDM 0
#endif

#if HAVE_TDM
typedef unsigned int uint32x4 __attribute__((ext_vector_type(4)));
typedef int          int32x4  __attribute__((ext_vector_type(4)));
typedef int          int32x8  __attribute__((ext_vector_type(8)));

// Issue one TDM 2-D tile load: bf16 elements, tile (tile_d0 x tile_d1) from a
// row-major tensor with row stride stride_d0 (elements) into LDS at `lds`.
// Descriptor layout per CDNA5 ISA sec.8 (D# groups 0/1; groups 2/3 zero = 2D).
__device__ __forceinline__ void tdm_load_2d(const void* gaddr, void* lds,
                                            unsigned tile_d0, unsigned tile_d1,
                                            unsigned tensor_d0, unsigned tensor_d1,
                                            unsigned stride_d0) {
  unsigned long long ga = (unsigned long long)(size_t)gaddr;
  unsigned la = (unsigned)(size_t)lds;
  uint32x4 g0;
  g0.x = 1u;                                   // count=1 (valid user D#)
  g0.y = la;                                   // lds_addr (bytes)
  g0.z = (unsigned)ga;                         // global_addr[31:0]
  g0.w = (unsigned)(ga >> 32) | (2u << 30);    // global_addr[56:32] | type=2
  int32x8 g1;
  g1[0] = (int)(1u << 16);                     // data_size=1 -> 2 bytes/elem
  g1[1] = (int)((tensor_d0 & 0xFFFFu) << 16);  // tensor_dim0[15:0] @ bit48
  g1[2] = (int)((tensor_d0 >> 16) | ((tensor_d1 & 0xFFFFu) << 16));
  g1[3] = (int)((tensor_d1 >> 16) | (tile_d0 << 16));   // tile_dim0 @ bit112
  g1[4] = (int)(tile_d1 & 0xFFFFu);            // tile_dim1 (tile_dim2 = 0)
  g1[5] = (int)stride_d0;                      // tensor_dim0_stride[31:0]
  g1[6] = 0;                                   // stride hi | dim1_stride lo
  g1[7] = 0;
  int32x4 gz = {0, 0, 0, 0};
#if __clang_major__ >= 23
  int32x8 gz8 = {0, 0, 0, 0, 0, 0, 0, 0};
  __builtin_amdgcn_tensor_load_to_lds(g0, g1, gz, gz, gz8, 0);
#else
  __builtin_amdgcn_tensor_load_to_lds(g0, g1, gz, gz, 0);
#endif
}
#endif  // HAVE_TDM

// float -> bf16 with round-to-nearest-even
__device__ __forceinline__ u16 f2bf(float f) {
  union { float f; unsigned u; } x; x.f = f;
  unsigned r = x.u + 0x7FFFu + ((x.u >> 16) & 1u);
  return (u16)(r >> 16);
}

// Build a 16-element bf16 fragment from two 16-byte halves (LDS or global).
__device__ __forceinline__ v16bf load_frag(const u16* p0, const u16* p1) {
  uint4 a = *reinterpret_cast<const uint4*>(p0);
  uint4 b = *reinterpret_cast<const uint4*>(p1);
  v16bf r;
  __builtin_memcpy(&r, &a, 16);
  __builtin_memcpy(reinterpret_cast<char*>(&r) + 16, &b, 16);
  return r;
}

__device__ __forceinline__ v8f wmma_bf16(v16bf a, v16bf b, v8f c) {
  // (neg_a, A, neg_b, B, c_mod, C, reuse_a, reuse_b)
  return __builtin_amdgcn_wmma_f32_16x16x32_bf16(false, a, false, b, (short)0, c,
                                                 false, false);
}

// ---------------------------------------------------------------------------
// fp32 -> bf16 (flat)
// ---------------------------------------------------------------------------
__global__ __launch_bounds__(256)
void cvt_f32_bf16(const float* __restrict__ src, u16* __restrict__ dst, int n) {
  int i = (blockIdx.x * 256 + threadIdx.x) * 4;
  if (i + 3 < n) {
    float4 v = *reinterpret_cast<const float4*>(src + i);
    ushort4 o;
    o.x = f2bf(v.x); o.y = f2bf(v.y); o.z = f2bf(v.z); o.w = f2bf(v.w);
    *reinterpret_cast<ushort4*>(dst + i) = o;
  }
}

// fp32 [K][N] -> bf16 [N][K]  (one-time weight transpose so WMMA B-fragments
// become contiguous 32B runs)
__global__ __launch_bounds__(256)
void cvtT_f32_bf16(const float* __restrict__ src, u16* __restrict__ dst,
                   int K, int N) {
  size_t i = (size_t)blockIdx.x * 256 + threadIdx.x;
  if (i < (size_t)K * N) {
    int k = (int)(i / N), n = (int)(i % N);
    dst[(size_t)n * K + k] = f2bf(src[i]);
  }
}

// ---------------------------------------------------------------------------
// Generic bf16 WMMA GEMM: C[M,N] = A[M,K] * Bt[N,K]^T (+bias, relu, residual)
// Block = 256 threads (8 wave32), tile 128x128, K-step 32.
// TDM path: wave 0 DMAs the next A/B tiles into the ping-pong LDS buffers
// while all waves compute on the current ones; one barrier per K-step.
// ---------------------------------------------------------------------------
template<bool RELU, bool RESID, bool OUTF, bool OUTB>
__global__ __launch_bounds__(256)
void gemm_bf16_wmma(const u16* __restrict__ A, const u16* __restrict__ Bt,
                    const float* __restrict__ bias,
                    const float* __restrict__ resid, int ldr,
                    float* __restrict__ Cf, u16* __restrict__ Cb,
                    int M, int N, int K) {
#if HAVE_TDM
  __shared__ __align__(16) u16 As[2][128 * 32];
  __shared__ __align__(16) u16 Bs[2][128 * 32];
#else
  __shared__ __align__(16) u16 As[1][128 * 32];
  __shared__ __align__(16) u16 Bs[1][128 * 32];
#endif

  const int t    = threadIdx.x;
  const int lane = t & 31;
  const int w    = t >> 5;
  const int mb   = blockIdx.y * 128;
  const int nb   = blockIdx.x * 128;

  v8f acc[8] = {};
  const int nsteps = K / 32;

#if HAVE_TDM
  const u16* gAt = A  + (size_t)mb * K;
  const u16* gBt = Bt + (size_t)nb * K;
  if (w == 0) {
    tdm_load_2d(gAt, &As[0][0], 32, 128, K, 128, K);
    tdm_load_2d(gBt, &Bs[0][0], 32, 128, K, 128, K);
  }
#else
  // manual staging map: thread t moves 16 bf16 (32B) of each tile per K-step
  const int srow = t >> 1;
  const int scol = (t & 1) * 16;
  const u16* gA = A  + (size_t)(mb + srow) * K + scol;
  const u16* gB = Bt + (size_t)(nb + srow) * K + scol;
  u16* sA = &As[0][0] + srow * 32 + scol;
  u16* sB = &Bs[0][0] + srow * 32 + scol;
#endif

  for (int i = 0; i < nsteps; ++i) {
    const int kb = i * 32;
#if HAVE_TDM
    const int cur = i & 1;
    __builtin_amdgcn_s_wait_tensorcnt(0);   // current buffers ready (wave 0)
    __syncthreads();                        // publish to all waves
    if (w == 0 && i + 1 < nsteps) {         // kick off next tiles async
      tdm_load_2d(gAt + (size_t)(kb + 32), &As[cur ^ 1][0], 32, 128, K, 128, K);
      tdm_load_2d(gBt + (size_t)(kb + 32), &Bs[cur ^ 1][0], 32, 128, K, 128, K);
    }
    const u16* Abuf = &As[cur][0];
    const u16* Bbuf = &Bs[cur][0];
#else
    {
      const uint4* ga = reinterpret_cast<const uint4*>(gA + kb);
      const uint4* gb = reinterpret_cast<const uint4*>(gB + kb);
      uint4 a0 = ga[0], a1 = ga[1];
      uint4 b0 = gb[0], b1 = gb[1];
      if (kb + 32 < K) {
        __builtin_prefetch(gA + kb + 32, 0, 1);
        __builtin_prefetch(gB + kb + 32, 0, 1);
      }
      reinterpret_cast<uint4*>(sA)[0] = a0;
      reinterpret_cast<uint4*>(sA)[1] = a1;
      reinterpret_cast<uint4*>(sB)[0] = b0;
      reinterpret_cast<uint4*>(sB)[1] = b1;
    }
    __syncthreads();
    const u16* Abuf = &As[0][0];
    const u16* Bbuf = &Bs[0][0];
#endif

    // A fragment: lane(0-15)=row m, K 0-7 & 16-23; lane(16-31)=row, K 8-15 & 24-31
    const int rowA = w * 16 + (lane & 15);
    const int ka   = (lane >> 4) * 8;
    const u16* pa  = Abuf + rowA * 32 + ka;
    v16bf afrag = load_frag(pa, pa + 16);

#pragma unroll
    for (int j = 0; j < 8; ++j) {
      // B fragment: lane&15 = output column, lane(0-15): K 0-7 & 16-23; lane(16-31): K 8-15 & 24-31
      const int rowB = j * 16 + (lane & 15);
      const int kbk  = (lane >> 4) * 8;
      const u16* pb  = Bbuf + rowB * 32 + kbk;
      v16bf bfrag = load_frag(pb, pb + 16);
      acc[j] = wmma_bf16(afrag, bfrag, acc[j]);
    }
#if !HAVE_TDM
    __syncthreads();
#endif
  }

  // Epilogue. C layout: VGPR r -> row (r + 8*(lane>=16)), col = lane&15.
#pragma unroll
  for (int j = 0; j < 8; ++j) {
    union { v8f v; float f[8]; } u; u.v = acc[j];
    const int col = nb + j * 16 + (lane & 15);
    const float bc = bias[col];
#pragma unroll
    for (int r = 0; r < 8; ++r) {
      const int row = mb + w * 16 + r + ((lane >> 4) << 3);
      float v = u.f[r] + bc;
      if constexpr (RELU)  v = fmaxf(v, 0.0f);
      if constexpr (RESID) v += resid[(size_t)row * ldr + col];
      if constexpr (OUTF)  Cf[(size_t)row * N + col] = v;
      if constexpr (OUTB)  Cb[(size_t)row * N + col] = f2bf(v);
    }
  }
}

// ---------------------------------------------------------------------------
// LayerNorm of q,k slices of kqv (split order k|q|v) + scatter to attention
// layouts: q,k -> [B,H,S,Dh] bf16 ; v -> [B,H,Dh,S] bf16 (pre-transposed).
// One row (b,s) per 256-thread block.
// ---------------------------------------------------------------------------
__global__ __launch_bounds__(256)
void ln_split_kernel(const float* __restrict__ kqv,
                     const float* __restrict__ gamma,
                     const float* __restrict__ beta,
                     u16* __restrict__ qo, u16* __restrict__ ko,
                     u16* __restrict__ vto) {
  __shared__ float red[256];
  const int t   = threadIdx.x;
  const int row = blockIdx.x;           // 0..MROWS-1
  const int b   = row / SEQ, s = row % SEQ;
  const float* base = kqv + (size_t)row * (3 * DMODEL);

  float kx[2], qx[2], vx[2];
  float ksum = 0.f, ksq = 0.f, qsum = 0.f, qsq = 0.f;
#pragma unroll
  for (int i = 0; i < 2; ++i) {
    int d = t + i * 256;
    kx[i] = base[d];
    qx[i] = base[DMODEL + d];
    vx[i] = base[2 * DMODEL + d];
    ksum += kx[i]; ksq += kx[i] * kx[i];
    qsum += qx[i]; qsq += qx[i] * qx[i];
  }

  auto bsum = [&](float v) -> float {
    red[t] = v; __syncthreads();
    for (int o = 128; o > 0; o >>= 1) {
      if (t < o) red[t] += red[t + o];
      __syncthreads();
    }
    float r = red[0]; __syncthreads();
    return r;
  };

  float kS = bsum(ksum), kQ = bsum(ksq);
  float qS = bsum(qsum), qQ = bsum(qsq);
  const float inv = 1.0f / (float)DMODEL;
  float km = kS * inv, kvr = kQ * inv - km * km;
  float qm = qS * inv, qvr = qQ * inv - qm * qm;
  float krs = rsqrtf(kvr + 1e-3f), qrs = rsqrtf(qvr + 1e-3f);

#pragma unroll
  for (int i = 0; i < 2; ++i) {
    int d = t + i * 256;
    int h = d >> 6, dh = d & 63;
    float g = gamma[d], be = beta[d];
    size_t bh = (size_t)(b * NHEADS + h);
    qo[(bh * SEQ + s) * HEADSZ + dh]   = f2bf((qx[i] - qm) * qrs * g + be);
    ko[(bh * SEQ + s) * HEADSZ + dh]   = f2bf((kx[i] - km) * krs * g + be);
    vto[(bh * HEADSZ + dh) * SEQ + s]  = f2bf(vx[i]);
  }
}

// ---------------------------------------------------------------------------
// Flash-style attention: grid = (B*H, SEQ/128). Block = 8 waves; each wave owns
// 16 query rows. 64-key K and V^T tiles staged in LDS (TDM double-buffered
// when available). Online softmax.
// ---------------------------------------------------------------------------
__global__ __launch_bounds__(256)
void attn_wmma_kernel(const u16* __restrict__ Q, const u16* __restrict__ Kl,
                      const u16* __restrict__ Vt, u16* __restrict__ Out) {
#if HAVE_TDM
  __shared__ __align__(16) u16 Ks[2][64 * 64];     // [key][dh]
  __shared__ __align__(16) u16 Vs[2][64 * 64];     // [dh][key] (pre-transposed V)
#else
  __shared__ __align__(16) u16 Ks[1][64 * 64];
  __shared__ __align__(16) u16 Vs[1][64 * 64];
#endif
  __shared__ __align__(16) u16 Ps[8 * 16 * 64];    // per-wave P tile [16][64]

  const int t = threadIdx.x, lane = t & 31, w = t >> 5;
  const int bh = blockIdx.x;                       // b*H + h
  const int b = bh >> 3, h = bh & 7;
  const int qb = blockIdx.y * 128;

  const u16* qbase = Q  + (size_t)bh * SEQ * HEADSZ;
  const u16* kbase = Kl + (size_t)bh * SEQ * HEADSZ;
  const u16* vbase = Vt + (size_t)bh * HEADSZ * SEQ;

  // Q fragments kept in registers for all key tiles (K-dim 64 = two chunks)
  const int rowQ = qb + w * 16 + (lane & 15);
  const int ka8  = (lane >> 4) * 8;
  v16bf qa[2];
#pragma unroll
  for (int c = 0; c < 2; ++c) {
    const u16* p = qbase + (size_t)rowQ * HEADSZ + c * 32 + ka8;
    qa[c] = load_frag(p, p + 16);
  }

  float m_i[8], l_i[8];
#pragma unroll
  for (int r = 0; r < 8; ++r) { m_i[r] = -3.0e38f; l_i[r] = 0.0f; }
  v8f oacc[4] = {};

  u16* pp = Ps + w * 16 * 64;
  const int nkt = SEQ / 64;

#if HAVE_TDM
  if (w == 0) {
    tdm_load_2d(kbase, &Ks[0][0], 64, 64, 64, SEQ, 64);
    tdm_load_2d(vbase, &Vs[0][0], 64, 64, SEQ, 64, SEQ);
  }
#else
  const int srow = t >> 2, scol = (t & 3) * 16;    // tile staging map
#endif

  for (int it = 0; it < nkt; ++it) {
    const int kt = it * 64;
#if HAVE_TDM
    const int cur = it & 1;
    __builtin_amdgcn_s_wait_tensorcnt(0);
    __syncthreads();
    if (w == 0 && it + 1 < nkt) {
      tdm_load_2d(kbase + (size_t)(kt + 64) * HEADSZ, &Ks[cur ^ 1][0],
                  64, 64, 64, SEQ, 64);
      tdm_load_2d(vbase + (kt + 64), &Vs[cur ^ 1][0],
                  64, 64, SEQ, 64, SEQ);
    }
    const u16* Kbuf = &Ks[cur][0];
    const u16* Vbuf = &Vs[cur][0];
#else
    {  // stage K tile (row-major [key][dh]) and V^T tile ([dh][key])
      const uint4* gk = reinterpret_cast<const uint4*>(
          kbase + (size_t)(kt + srow) * HEADSZ + scol);
      const uint4* gv = reinterpret_cast<const uint4*>(
          vbase + (size_t)srow * SEQ + kt + scol);
      uint4* sk = reinterpret_cast<uint4*>(&Ks[0][0] + srow * 64 + scol);
      uint4* sv = reinterpret_cast<uint4*>(&Vs[0][0] + srow * 64 + scol);
      sk[0] = gk[0]; sk[1] = gk[1];
      sv[0] = gv[0]; sv[1] = gv[1];
    }
    __syncthreads();
    const u16* Kbuf = &Ks[0][0];
    const u16* Vbuf = &Vs[0][0];
#endif

    // S = Q @ K^T : 4 key sub-tiles x 2 K-chunks = 8 WMMAs -> 16x64 scores
    v8f sc[4] = {};
#pragma unroll
    for (int c = 0; c < 2; ++c) {
#pragma unroll
      for (int n = 0; n < 4; ++n) {
        const u16* pb = Kbuf + (n * 16 + (lane & 15)) * 64 + c * 32 + (lane >> 4) * 8;
        sc[n] = wmma_bf16(qa[c], load_frag(pb, pb + 16), sc[n]);
      }
    }

    float sf[4][8];
#pragma unroll
    for (int n = 0; n < 4; ++n) {
      union { v8f v; float f[8]; } u; u.v = sc[n];
#pragma unroll
      for (int r = 0; r < 8; ++r) sf[n][r] = u.f[r] * 0.125f;  // 1/sqrt(64)
    }

    // online softmax; row lives in 16 lanes (xor 1,2,4,8 stays in half-wave)
    float alpha_r[8];
#pragma unroll
    for (int r = 0; r < 8; ++r) {
      float mx = fmaxf(fmaxf(sf[0][r], sf[1][r]), fmaxf(sf[2][r], sf[3][r]));
#pragma unroll
      for (int o = 8; o >= 1; o >>= 1) mx = fmaxf(mx, __shfl_xor(mx, o, 32));
      float mnew  = fmaxf(m_i[r], mx);
      float alpha = __expf(m_i[r] - mnew);
      float ps = 0.0f;
#pragma unroll
      for (int n = 0; n < 4; ++n) {
        float p = __expf(sf[n][r] - mnew);
        sf[n][r] = p;
        ps += p;
      }
#pragma unroll
      for (int o = 8; o >= 1; o >>= 1) ps += __shfl_xor(ps, o, 32);
      l_i[r] = l_i[r] * alpha + ps;
      m_i[r] = mnew;
      alpha_r[r] = alpha;
    }
#pragma unroll
    for (int j = 0; j < 4; ++j) {
      union { v8f v; float f[8]; } u; u.v = oacc[j];
#pragma unroll
      for (int r = 0; r < 8; ++r) u.f[r] *= alpha_r[r];
      oacc[j] = u.v;
    }

    // P (C-layout) -> LDS row-major so it can be reloaded as an A operand
#pragma unroll
    for (int n = 0; n < 4; ++n)
#pragma unroll
      for (int r = 0; r < 8; ++r)
        pp[(r + ((lane >> 4) << 3)) * 64 + n * 16 + (lane & 15)] = f2bf(sf[n][r]);

    // O += P @ V : 2 key chunks x 4 dh sub-tiles = 8 WMMAs
#pragma unroll
    for (int c = 0; c < 2; ++c) {
      const u16* pa = pp + (lane & 15) * 64 + c * 32 + ka8;
      v16bf pfrag = load_frag(pa, pa + 16);
#pragma unroll
      for (int j = 0; j < 4; ++j) {
        const u16* pb = Vbuf + (j * 16 + (lane & 15)) * 64 + c * 32 + (lane >> 4) * 8;
        oacc[j] = wmma_bf16(pfrag, load_frag(pb, pb + 16), oacc[j]);
      }
    }
#if !HAVE_TDM
    __syncthreads();
#endif
  }

  // normalize and write [B*S, D] bf16 (col = h*64 + dh)
#pragma unroll
  for (int j = 0; j < 4; ++j) {
    union { v8f v; float f[8]; } u; u.v = oacc[j];
#pragma unroll
    for (int r = 0; r < 8; ++r) {
      int row = qb + w * 16 + r + ((lane >> 4) << 3);
      int col = h * HEADSZ + j * 16 + (lane & 15);
      Out[((size_t)b * SEQ + row) * DMODEL + col] = f2bf(u.f[r] / l_i[r]);
    }
  }
}

// ---------------------------------------------------------------------------
// Host-side orchestration
// ---------------------------------------------------------------------------
extern "C" void kernel_launch(void* const* d_in, const int* in_sizes, int n_in,
                              void* d_out, int out_size, void* d_ws, size_t ws_size,
                              hipStream_t stream) {
  (void)in_sizes; (void)n_in; (void)out_size; (void)ws_size;
  const float* values   = (const float*)d_in[0];
  const float* kqv_w    = (const float*)d_in[1];
  const float* kqv_b    = (const float*)d_in[2];
  const float* ln_gamma = (const float*)d_in[3];
  const float* ln_beta  = (const float*)d_in[4];
  const float* kernel_w = (const float*)d_in[5];
  const float* kernel_b = (const float*)d_in[6];
  const float* proj_w   = (const float*)d_in[7];
  const float* proj_b   = (const float*)d_in[8];
  float* out = (float*)d_out;

  char* ws = (char*)d_ws;
  size_t off = 0;
  auto take = [&](size_t bytes) -> void* {
    void* p = ws + off;
    off = (off + bytes + 255) & ~(size_t)255;
    return p;
  };

  u16*   values_bf = (u16*)  take((size_t)MROWS * DMODEL * 2);
  u16*   wqkvT     = (u16*)  take((size_t)(3 * DMODEL) * DMODEL * 2);
  u16*   wkerT     = (u16*)  take((size_t)KSIZE * DMODEL * 2);
  u16*   wprojT    = (u16*)  take((size_t)DMODEL * KSIZE * 2);
  float* kqv       = (float*)take((size_t)MROWS * 3 * DMODEL * 4);
  u16*   qln       = (u16*)  take((size_t)MROWS * DMODEL * 2);
  u16*   kln       = (u16*)  take((size_t)MROWS * DMODEL * 2);
  u16*   vt        = (u16*)  take((size_t)MROWS * DMODEL * 2);
  u16*   attn      = (u16*)  take((size_t)MROWS * DMODEL * 2);
  u16*   hidden    = (u16*)  take((size_t)MROWS * KSIZE * 2);

  const dim3 blk(256);

  // 1) precision conversions (+ weight transposes)
  {
    int n = MROWS * DMODEL;
    cvt_f32_bf16<<<n / (256 * 4), blk, 0, stream>>>(values, values_bf, n);
  }
  cvtT_f32_bf16<<<(unsigned)(((size_t)DMODEL * 3 * DMODEL + 255) / 256), blk, 0, stream>>>(
      kqv_w, wqkvT, DMODEL, 3 * DMODEL);
  cvtT_f32_bf16<<<(unsigned)(((size_t)DMODEL * KSIZE + 255) / 256), blk, 0, stream>>>(
      kernel_w, wkerT, DMODEL, KSIZE);
  cvtT_f32_bf16<<<(unsigned)(((size_t)KSIZE * DMODEL + 255) / 256), blk, 0, stream>>>(
      proj_w, wprojT, KSIZE, DMODEL);

  // 2) kqv = values @ kqv_w + kqv_b   (f32 out, needed for LN + v residual)
  gemm_bf16_wmma<false, false, true, false>
      <<<dim3((3 * DMODEL) / 128, MROWS / 128), blk, 0, stream>>>(
          values_bf, wqkvT, kqv_b, nullptr, 0, kqv, nullptr,
          MROWS, 3 * DMODEL, DMODEL);

  // 3) layernorm q,k + scatter to attention layouts (split order: k|q|v)
  ln_split_kernel<<<MROWS, blk, 0, stream>>>(kqv, ln_gamma, ln_beta, qln, kln, vt);

  // 4) attention -> bf16 [B*S, D]
  attn_wmma_kernel<<<dim3(BATCH * NHEADS, SEQ / 128), blk, 0, stream>>>(
      qln, kln, vt, attn);

  // 5) hidden = relu(attn @ kernel_w + kernel_b)  (bf16 out)
  gemm_bf16_wmma<true, false, false, true>
      <<<dim3(KSIZE / 128, MROWS / 128), blk, 0, stream>>>(
          attn, wkerT, kernel_b, nullptr, 0, nullptr, hidden,
          MROWS, KSIZE, DMODEL);

  // 6) out = hidden @ proj_w + proj_b + v_residual  (v = kqv cols 1024..1535)
  gemm_bf16_wmma<false, true, true, false>
      <<<dim3(DMODEL / 128, MROWS / 128), blk, 0, stream>>>(
          hidden, wprojT, proj_b, kqv + 2 * DMODEL, 3 * DMODEL, out, nullptr,
          MROWS, DMODEL, KSIZE);
}
